// GRUModel_47725676593782
// MI455X (gfx1250) — matrix operand order repaired and checked by reading, not compiled, not run on hardware
//
#include <hip/hip_runtime.h>

// GRU seq2seq for MI455X (gfx1250, wave32, WMMA).
// Strategy: hoist input-gate GEMMs + attention projection into big parallel
// bf16 WMMA GEMMs; run the sequential recurrences as persistent per-(slice,dir)
// workgroups with h resident in LDS; weights pre-packed into WMMA B-fragment
// layout (one 32B load per lane per fragment). All matmul loops are
// kt-outer / nt-inner so each A fragment is loaded+converted once and reused
// across 2-8 back-to-back v_wmma issues.

#define T_ 256
#define B_ 64
#define H_ 256
#define D_ 512
#define G3H 768
#define G3D 1536
#define NSTEPS 24

typedef __bf16 v16bf __attribute__((ext_vector_type(16)));
typedef float v8f __attribute__((ext_vector_type(8)));

static __device__ inline v8f wmma_bf16(v16bf a, v16bf b, v8f c) {
  // D = A(16x32) * B(32x16) + C, f32 accumulate
  return __builtin_amdgcn_wmma_f32_16x16x32_bf16(false, a, false, b, (short)0, c,
                                                 false, false);
}

// A-fragment (16x32, M x K) per ISA 7.12.2 16-bit A layout:
// lane = half*16 + m ; K(v,half,j) = (v>>2)*16 + half*8 + (v&3)*2 + j
static __device__ inline v16bf load_a_frag_f32(const float* A, int lda, int m0, int k0) {
  int lane = threadIdx.x & 31;
  const float* base = A + (size_t)(m0 + (lane & 15)) * lda + k0 + ((lane >> 4) << 3);
  v16bf out;
#pragma unroll
  for (int v = 0; v < 8; ++v) {
    int k = ((v >> 2) << 4) + ((v & 3) << 1);
    out[2 * v] = (__bf16)base[k];
    out[2 * v + 1] = (__bf16)base[k + 1];
  }
  return out;
}

static __device__ inline v16bf load_a_frag_bf16(const __bf16* A, int lda, int m0, int k0) {
  int lane = threadIdx.x & 31;
  const __bf16* base = A + (size_t)(m0 + (lane & 15)) * lda + k0 + ((lane >> 4) << 3);
  v16bf out;
#pragma unroll
  for (int v = 0; v < 8; ++v) {
    int k = ((v >> 2) << 4) + ((v & 3) << 1);
    out[2 * v] = base[k];
    out[2 * v + 1] = base[k + 1];
  }
  return out;
}

// B operand is pre-packed per (nt,kt) tile: 512 bf16 per tile, lane-major, so a
// fragment is one contiguous 32-byte load per lane.
static __device__ inline v16bf load_b_packed(const __bf16* P, int ktiles, int nt, int kt) {
  const __bf16* ptr = P + (((size_t)nt * ktiles + kt) << 9) + ((threadIdx.x & 31) << 4);
  return *(const v16bf*)ptr;
}

// C/D fragment: row = m0 + half*8 + r, col = n0 + (lane&15)
static __device__ inline void store_c_f32(float* C, int ldc, int m0, int n0, v8f c) {
  int lane = threadIdx.x & 31;
  float* base = C + (size_t)(m0 + ((lane >> 4) << 3)) * ldc + n0 + (lane & 15);
#pragma unroll
  for (int r = 0; r < 8; ++r) base[(size_t)r * ldc] = c[r];
}

static __device__ inline void store_c_bf16(__bf16* C, int ldc, int m0, int n0, v8f c) {
  int lane = threadIdx.x & 31;
  __bf16* base = C + (size_t)(m0 + ((lane >> 4) << 3)) * ldc + n0 + (lane & 15);
#pragma unroll
  for (int r = 0; r < 8; ++r) base[(size_t)r * ldc] = (__bf16)c[r];
}

static __device__ inline float sigmoidf_(float x) { return 1.0f / (1.0f + __expf(-x)); }

// ---------------------------------------------------------------------------
// Weight prep: f32 (N x K, row stride srcLd) -> bf16 packed B-fragment tiles.
// B-fragment 32x16 (KxN) 16-bit layout: lane = khalf*16 + n,
// K(e,khalf) = (e>>3)*16 + khalf*8 + (e&7)  (matches A fragment order).
__global__ __launch_bounds__(256) void k_pack_b(const float* __restrict__ W,
                                                __bf16* __restrict__ P, int N, int K,
                                                int srcLd, size_t srcStrideZ,
                                                size_t dstStrideZ) {
  size_t idx = (size_t)blockIdx.x * 256 + threadIdx.x;
  size_t total = (size_t)N * K;
  if (idx >= total) return;
  const float* Wz = W + (size_t)blockIdx.z * srcStrideZ;
  __bf16* Pz = P + (size_t)blockIdx.z * dstStrideZ;
  int ktiles = K >> 5;
  int e = (int)(idx & 15);
  int lane = (int)((idx >> 4) & 31);
  int tile = (int)(idx >> 9);
  int kt = tile % ktiles, nt = tile / ktiles;
  int v = e >> 1, j = e & 1;
  int k = (kt << 5) + ((lane >> 4) << 3) + ((v >> 2) << 4) + ((v & 3) << 1) + j;
  int n = (nt << 4) + (lane & 15);
  Pz[idx] = (__bf16)Wz[(size_t)n * srcLd + k];
}

// ---------------------------------------------------------------------------
// Layer-0 input gates (K=2 is degenerate; pure VALU).
// GI layout (dir, t, b, g) flattened.
__global__ __launch_bounds__(256) void k_gi0(const float* __restrict__ x,
                                             const float* __restrict__ Wih0,
                                             const float* __restrict__ bih0,
                                             float* __restrict__ GI) {
  size_t idx = (size_t)blockIdx.x * 256 + threadIdx.x;
  int g = (int)(idx % G3H);
  size_t r = idx / G3H;
  int b = (int)(r % B_);
  size_t r2 = r / B_;
  int t = (int)(r2 % T_);
  int dir = (int)(r2 / T_);
  const float* xv = x + ((size_t)b * T_ + t) * 2;
  const float* w = Wih0 + ((size_t)dir * G3H + g) * 2;
  GI[idx] = xv[0] * w[0] + xv[1] * w[1] + bih0[dir * G3H + g];
}

// ---------------------------------------------------------------------------
// Generic big GEMM: C(MxN,f32) = cvt_bf16(A) * Bpacked + bias[n].
// Each wave computes a 16x32 strip (two N-tiles): A fragment loaded once per
// K-step and reused by two WMMAs. 8 waves/block along N, grid.z batches dirs.
__global__ __launch_bounds__(256) void k_gemm_bias(
    const float* __restrict__ A, int lda, size_t aStrideZ,
    const __bf16* __restrict__ Bp, size_t bStrideZ,
    const float* __restrict__ bias, size_t biasStrideZ,
    float* __restrict__ C, int ldc, size_t cStrideZ, int ktiles) {
  int z = blockIdx.z;
  const float* Az = A + (size_t)z * aStrideZ;
  const __bf16* Bz = Bp + (size_t)z * bStrideZ;
  const float* bz = bias + (size_t)z * biasStrideZ;
  float* Cz = C + (size_t)z * cStrideZ;
  int wave = threadIdx.x >> 5;
  int lane = threadIdx.x & 31;
  int m0 = blockIdx.x << 4;
  int nt0 = (blockIdx.y * 8 + wave) * 2;
  v8f acc0, acc1;
  {
    float bv0 = bz[(nt0 << 4) + (lane & 15)];
    float bv1 = bz[((nt0 + 1) << 4) + (lane & 15)];
#pragma unroll
    for (int r = 0; r < 8; ++r) { acc0[r] = bv0; acc1[r] = bv1; }
  }
  for (int kt = 0; kt < ktiles; ++kt) {
    if (kt + 1 < ktiles) {
      __builtin_prefetch(Bz + (((size_t)nt0 * ktiles + kt + 1) << 9), 0, 1);
      __builtin_prefetch(Bz + (((size_t)(nt0 + 1) * ktiles + kt + 1) << 9), 0, 1);
    }
    v16bf a = load_a_frag_f32(Az, lda, m0, kt << 5);
    acc0 = wmma_bf16(a, load_b_packed(Bz, ktiles, nt0, kt), acc0);
    acc1 = wmma_bf16(a, load_b_packed(Bz, ktiles, nt0 + 1, kt), acc1);
  }
  store_c_f32(Cz, ldc, m0, nt0 << 4, acc0);
  store_c_f32(Cz, ldc, m0, (nt0 + 1) << 4, acc1);
}

// ---------------------------------------------------------------------------
// Encoder recurrence: one workgroup per (batch-slice of 16, direction),
// h resident in LDS across all T steps; per step all 8 A fragments of h are
// hoisted into VGPRs once, then 6 N-tiles of 8 WMMAs stream B from L2.
__global__ __launch_bounds__(256) void k_enc_recur(
    const float* __restrict__ GI,      // (2, T*B, 768)
    const __bf16* __restrict__ WhhP,   // (2) packed ktiles=8, ntiles=48
    const float* __restrict__ bhh,     // (2, 768)
    float* __restrict__ seq_out,       // (T, B, 512)
    float* __restrict__ hlast)         // (2, B, 256)
{
  __shared__ float h[16 * H_];
  __shared__ float gh[16 * G3H];
  int row0 = blockIdx.x * 16;
  int dir = blockIdx.y;
  const float* GId = GI + (size_t)dir * T_ * B_ * G3H;
  const __bf16* Wd = WhhP + (size_t)dir * (H_ * G3H);
  const float* bd = bhh + dir * G3H;
  int tid = threadIdx.x, wave = tid >> 5, lane = tid & 31;
  for (int i = tid; i < 16 * H_; i += 256) h[i] = 0.f;
  __syncthreads();
  for (int step = 0; step < T_; ++step) {
    int t = dir ? (T_ - 1 - step) : step;
    // hoist all 8 A fragments of h (64 VGPRs), reused by all 6 N-tiles
    v16bf afr[8];
#pragma unroll
    for (int kt = 0; kt < 8; ++kt) afr[kt] = load_a_frag_f32(h, H_, 0, kt << 5);
#pragma unroll 1
    for (int nt = wave; nt < 48; nt += 8) {
      int n0 = nt << 4;
      float bv = bd[n0 + (lane & 15)];
      v8f acc;
#pragma unroll
      for (int r = 0; r < 8; ++r) acc[r] = bv;
#pragma unroll
      for (int kt = 0; kt < 8; ++kt)
        acc = wmma_bf16(afr[kt], load_b_packed(Wd, 8, nt, kt), acc);
      store_c_f32(gh, G3H, 0, n0, acc);
    }
    __syncthreads();
    const float* gi = GId + ((size_t)t * B_ + row0) * G3H;
#pragma unroll 1
    for (int i = tid; i < 16 * H_; i += 256) {
      int b = i >> 8, j = i & (H_ - 1);
      const float* gib = gi + (size_t)b * G3H;
      const float* ghb = gh + b * G3H;
      float rr = sigmoidf_(gib[j] + ghb[j]);
      float zz = sigmoidf_(gib[H_ + j] + ghb[H_ + j]);
      float nn = tanhf(gib[2 * H_ + j] + rr * ghb[2 * H_ + j]);
      float hv = (1.f - zz) * nn + zz * h[i];
      h[i] = hv;
      seq_out[((size_t)t * B_ + row0 + b) * D_ + dir * H_ + j] = hv;
    }
    __syncthreads();
  }
  for (int i = tid; i < 16 * H_; i += 256) {
    int b = i >> 8, j = i & (H_ - 1);
    hlast[((size_t)dir * B_ + row0 + b) * H_ + j] = h[i];
  }
}

// ---------------------------------------------------------------------------
// Decoder: 4 workgroups (16 batch rows each), 24 sequential steps; hidden
// stack + all step state in LDS (~258KB of 320KB/WGP). All matmuls kt-outer:
// one A fragment live, 4-8 accumulators per wave.
__global__ __launch_bounds__(256) void k_decoder(
    const float* __restrict__ enc,     // (T,B,512) encoder output
    const float* __restrict__ proj,    // (T,B,512) enc_proj + attn_b
    const float* __restrict__ hlast,   // (3,2,B,256)
    const __bf16* __restrict__ WqP,    // attn_Wh packed kt=16 nt=32
    const __bf16* __restrict__ WcP,    // fc_con packed kt=32 nt=32
    const float* __restrict__ bc,      // fc_con_b
    const __bf16* __restrict__ WihP,   // (3) packed kt=16 nt=96
    const __bf16* __restrict__ WhhP,   // (3) packed kt=16 nt=96
    const float* __restrict__ bih,     // (3,1536)
    const float* __restrict__ bhh,     // (3,1536)
    const float* __restrict__ vvec,    // (512)
    const float* __restrict__ fcW,     // (2,512)
    const float* __restrict__ fcb,     // (2)
    float* __restrict__ out)           // (B,24,2)
{
  __shared__ float hbuf[3][16 * D_];
  __shared__ float hq[16 * D_];
  __shared__ float aw[16 * T_];
  __shared__ __bf16 din_bf[16 * D_];
  __shared__ __bf16 inp_bf[16 * D_];
  __shared__ __bf16 ctx_bf[16 * D_];
  __shared__ __bf16 rz_bf[16 * 2 * D_];
  __shared__ __bf16 inn_bf[16 * D_];
  __shared__ __bf16 hn_bf[16 * D_];
  __shared__ float vsh[D_];

  int tid = threadIdx.x, wave = tid >> 5, lane = tid & 31;
  int row0 = blockIdx.x * 16;

  for (int i = tid; i < 3 * 16 * D_; i += 256) {
    int l = i / (16 * D_);
    int r = i - l * (16 * D_);
    int b = r >> 9, j = r & (D_ - 1);
    int dir = j >> 8, jj = j & (H_ - 1);
    hbuf[l][r] = hlast[(((size_t)l * 2 + dir) * B_ + row0 + b) * H_ + jj];
  }
  for (int i = tid; i < 16 * D_; i += 256) {
    int b = i >> 9, j = i & (D_ - 1);
    din_bf[i] = (__bf16)enc[((size_t)(T_ - 1) * B_ + row0 + b) * D_ + j];
  }
  for (int i = tid; i < D_; i += 256) vsh[i] = vvec[i];
  __syncthreads();

  for (int step = 0; step < NSTEPS; ++step) {
    // A: hq = h[2] @ attn_Wh^T  (wave: 4 N-tiles, kt-outer A reuse)
    {
      v8f acc[4];
#pragma unroll
      for (int i = 0; i < 4; ++i)
#pragma unroll
        for (int r = 0; r < 8; ++r) acc[i][r] = 0.f;
      for (int kt = 0; kt < 16; ++kt) {
        v16bf a = load_a_frag_f32(hbuf[2], D_, 0, kt << 5);
#pragma unroll
        for (int i = 0; i < 4; ++i)
          acc[i] = wmma_bf16(a, load_b_packed(WqP, 16, wave + 8 * i, kt), acc[i]);
      }
#pragma unroll
      for (int i = 0; i < 4; ++i)
        store_c_f32(hq, D_, 0, (wave + 8 * i) << 4, acc[i]);
    }
    __syncthreads();
    // B: scores[b][t] = v . tanh(hq + proj)
#pragma unroll 1
    for (int i = tid; i < 16 * T_; i += 256) {
      int t = i >> 4, b = i & 15;
      const float* P = proj + ((size_t)t * B_ + row0 + b) * D_;
      const float* hb = hq + b * D_;
      float s = 0.f;
      for (int d = 0; d < D_; ++d) s += vsh[d] * tanhf(hb[d] + P[d]);
      aw[b * T_ + t] = s;
    }
    __syncthreads();
    // C: softmax rows
    if (tid < 16) {
      float* r = aw + tid * T_;
      float mx = r[0];
      for (int t = 1; t < T_; ++t) mx = fmaxf(mx, r[t]);
      float s = 0.f;
      for (int t = 0; t < T_; ++t) { float e = __expf(r[t] - mx); r[t] = e; s += e; }
      float inv = 1.f / s;
      for (int t = 0; t < T_; ++t) r[t] *= inv;
    }
    __syncthreads();
    // D: ctx = aw @ enc
    {
      int b = tid >> 4;
      int d0 = (tid & 15) << 5;
      float acc[32];
#pragma unroll
      for (int q = 0; q < 32; ++q) acc[q] = 0.f;
#pragma unroll 1
      for (int t = 0; t < T_; ++t) {
        float w = aw[b * T_ + t];
        const float* e = enc + ((size_t)t * B_ + row0 + b) * D_ + d0;
#pragma unroll
        for (int q = 0; q < 32; ++q) acc[q] += w * e[q];
      }
#pragma unroll
      for (int q = 0; q < 32; ++q) ctx_bf[b * D_ + d0 + q] = (__bf16)acc[q];
    }
    __syncthreads();
    // E: inp = [din, ctx] @ fc_con_W^T + bc (split-K over the concat)
    {
      v8f acc[4];
#pragma unroll
      for (int i = 0; i < 4; ++i) {
        float bv = bc[((wave + 8 * i) << 4) + (lane & 15)];
#pragma unroll
        for (int r = 0; r < 8; ++r) acc[i][r] = bv;
      }
      for (int kt = 0; kt < 16; ++kt) {
        v16bf a = load_a_frag_bf16(din_bf, D_, 0, kt << 5);
#pragma unroll
        for (int i = 0; i < 4; ++i)
          acc[i] = wmma_bf16(a, load_b_packed(WcP, 32, wave + 8 * i, kt), acc[i]);
      }
      for (int kt = 0; kt < 16; ++kt) {
        v16bf a = load_a_frag_bf16(ctx_bf, D_, 0, kt << 5);
#pragma unroll
        for (int i = 0; i < 4; ++i)
          acc[i] = wmma_bf16(a, load_b_packed(WcP, 32, wave + 8 * i, 16 + kt), acc[i]);
      }
#pragma unroll
      for (int i = 0; i < 4; ++i)
        store_c_bf16(inp_bf, D_, 0, (wave + 8 * i) << 4, acc[i]);
    }
    __syncthreads();
    // F: 3 stacked GRU cells (D=512). Wave tiles: i=0..7 -> r,z (fused gi+gh),
    // i=8..11 -> n gate (separate inn / hn accumulators).
    for (int l = 0; l < 3; ++l) {
      const __bf16* Wi = WihP + (size_t)l * (D_ * G3D);
      const __bf16* Wh = WhhP + (size_t)l * (D_ * G3D);
      const float* bi = bih + l * G3D;
      const float* bh = bhh + l * G3D;
      float* hl = hbuf[l];
      const float* hprev = (l > 0) ? hbuf[l - 1] : hbuf[0];
      {
        // --- r,z gates ---
        v8f acc[8];
#pragma unroll
        for (int i = 0; i < 8; ++i) {
          int nl = ((wave + 8 * i) << 4) + (lane & 15);
          float bv = bi[nl] + bh[nl];
#pragma unroll
          for (int r = 0; r < 8; ++r) acc[i][r] = bv;
        }
        for (int kt = 0; kt < 16; ++kt) {
          v16bf a = (l == 0) ? load_a_frag_bf16(inp_bf, D_, 0, kt << 5)
                             : load_a_frag_f32(hprev, D_, 0, kt << 5);
#pragma unroll
          for (int i = 0; i < 8; ++i)
            acc[i] = wmma_bf16(a, load_b_packed(Wi, 16, wave + 8 * i, kt), acc[i]);
        }
        for (int kt = 0; kt < 16; ++kt) {
          v16bf a = load_a_frag_f32(hl, D_, 0, kt << 5);
#pragma unroll
          for (int i = 0; i < 8; ++i)
            acc[i] = wmma_bf16(a, load_b_packed(Wh, 16, wave + 8 * i, kt), acc[i]);
        }
#pragma unroll
        for (int i = 0; i < 8; ++i)
          store_c_bf16(rz_bf, 2 * D_, 0, (wave + 8 * i) << 4, acc[i]);
      }
      {
        // --- n gate: inn and hn kept separate ---
        v8f ai[4], ah[4];
#pragma unroll
        for (int i = 0; i < 4; ++i) {
          int nl = ((wave + 64 + 8 * i) << 4) + (lane & 15);
          float b0 = bi[nl], b1 = bh[nl];
#pragma unroll
          for (int r = 0; r < 8; ++r) { ai[i][r] = b0; ah[i][r] = b1; }
        }
        for (int kt = 0; kt < 16; ++kt) {
          v16bf a = (l == 0) ? load_a_frag_bf16(inp_bf, D_, 0, kt << 5)
                             : load_a_frag_f32(hprev, D_, 0, kt << 5);
#pragma unroll
          for (int i = 0; i < 4; ++i)
            ai[i] = wmma_bf16(a, load_b_packed(Wi, 16, wave + 64 + 8 * i, kt), ai[i]);
        }
        for (int kt = 0; kt < 16; ++kt) {
          v16bf a = load_a_frag_f32(hl, D_, 0, kt << 5);
#pragma unroll
          for (int i = 0; i < 4; ++i)
            ah[i] = wmma_bf16(a, load_b_packed(Wh, 16, wave + 64 + 8 * i, kt), ah[i]);
        }
#pragma unroll
        for (int i = 0; i < 4; ++i) {
          store_c_bf16(inn_bf, D_, 0, (wave + 8 * i) << 4, ai[i]);
          store_c_bf16(hn_bf, D_, 0, (wave + 8 * i) << 4, ah[i]);
        }
      }
      __syncthreads();
#pragma unroll 1
      for (int i = tid; i < 16 * D_; i += 256) {
        int b = i >> 9, j = i & (D_ - 1);
        float rr = sigmoidf_((float)rz_bf[b * 2 * D_ + j]);
        float zz = sigmoidf_((float)rz_bf[b * 2 * D_ + D_ + j]);
        float nn = tanhf((float)inn_bf[i] + rr * (float)hn_bf[i]);
        float hv = (1.f - zz) * nn + zz * hl[i];
        hl[i] = hv;
        if (l == 2) din_bf[i] = (__bf16)hv;  // d_in for next step
      }
      __syncthreads();
    }
    // G: out_step = h[2] @ fc_W^T + fc_b  (OUT=2, trivial)
    if (tid < 32) {
      int b = tid >> 1, o = tid & 1;
      const float* w = fcW + o * D_;
      const float* hb = hbuf[2] + b * D_;
      float s = fcb[o];
      for (int d = 0; d < D_; ++d) s += hb[d] * w[d];
      out[((size_t)(row0 + b) * NSTEPS + step) * 2 + o] = s;
    }
    __syncthreads();
  }
}

// ---------------------------------------------------------------------------
extern "C" void kernel_launch(void* const* d_in, const int* in_sizes, int n_in,
                              void* d_out, int out_size, void* d_ws, size_t ws_size,
                              hipStream_t stream) {
  (void)in_sizes; (void)n_in; (void)out_size; (void)ws_size;
  const float* x      = (const float*)d_in[0];
  const float* eWih0  = (const float*)d_in[1];
  const float* eWhh0  = (const float*)d_in[2];
  const float* ebih0  = (const float*)d_in[3];
  const float* ebhh0  = (const float*)d_in[4];
  const float* eWih   = (const float*)d_in[5];
  const float* eWhh   = (const float*)d_in[6];
  const float* ebih   = (const float*)d_in[7];
  const float* ebhh   = (const float*)d_in[8];
  const float* dWih   = (const float*)d_in[9];
  const float* dWhh   = (const float*)d_in[10];
  const float* dbih   = (const float*)d_in[11];
  const float* dbhh   = (const float*)d_in[12];
  const float* fcConW = (const float*)d_in[13];
  const float* fcConB = (const float*)d_in[14];
  const float* attnW  = (const float*)d_in[15];
  const float* attnB  = (const float*)d_in[16];
  const float* vvec   = (const float*)d_in[17];
  const float* fcW    = (const float*)d_in[18];
  const float* fcb    = (const float*)d_in[19];
  float* outp = (float*)d_out;

  char* p = (char*)d_ws;
  auto take = [&](size_t bytes) {
    char* r = p;
    p += (bytes + 255) & ~(size_t)255;
    return r;
  };
  __bf16* whhP    = (__bf16*)take((size_t)6 * H_ * G3H * 2);
  __bf16* wihP    = (__bf16*)take((size_t)4 * D_ * G3H * 2);
  __bf16* dWihP   = (__bf16*)take((size_t)3 * D_ * G3D * 2);
  __bf16* dWhhP   = (__bf16*)take((size_t)3 * D_ * G3D * 2);
  __bf16* fcConP  = (__bf16*)take((size_t)2 * D_ * D_ * 2);
  __bf16* attnWhP = (__bf16*)take((size_t)D_ * D_ * 2);
  __bf16* attnWeP = (__bf16*)take((size_t)D_ * D_ * 2);
  float* GIbuf    = (float*)take((size_t)2 * T_ * B_ * G3H * 4);
  float* seq1     = (float*)take((size_t)T_ * B_ * D_ * 4);
  float* seq2     = (float*)take((size_t)T_ * B_ * D_ * 4);
  float* projBuf  = (float*)take((size_t)T_ * B_ * D_ * 4);
  float* hlastBuf = (float*)take((size_t)3 * 2 * B_ * H_ * 4);

  dim3 blk(256);
  // ---- pack weights to bf16 fragment layout (once per call) ----
  k_pack_b<<<dim3((G3H * H_) / 256, 1, 2), blk, 0, stream>>>(
      eWhh0, whhP, G3H, H_, H_, (size_t)G3H * H_, (size_t)G3H * H_);
  k_pack_b<<<dim3((G3H * H_) / 256, 1, 4), blk, 0, stream>>>(
      eWhh, whhP + (size_t)2 * G3H * H_, G3H, H_, H_, (size_t)G3H * H_, (size_t)G3H * H_);
  k_pack_b<<<dim3((G3H * D_) / 256, 1, 4), blk, 0, stream>>>(
      eWih, wihP, G3H, D_, D_, (size_t)G3H * D_, (size_t)G3H * D_);
  k_pack_b<<<dim3((G3D * D_) / 256, 1, 3), blk, 0, stream>>>(
      dWih, dWihP, G3D, D_, D_, (size_t)G3D * D_, (size_t)G3D * D_);
  k_pack_b<<<dim3((G3D * D_) / 256, 1, 3), blk, 0, stream>>>(
      dWhh, dWhhP, G3D, D_, D_, (size_t)G3D * D_, (size_t)G3D * D_);
  k_pack_b<<<dim3((D_ * 2 * D_) / 256, 1, 1), blk, 0, stream>>>(
      fcConW, fcConP, D_, 2 * D_, 2 * D_, 0, 0);
  k_pack_b<<<dim3((D_ * D_) / 256, 1, 1), blk, 0, stream>>>(
      attnW, attnWhP, D_, D_, 2 * D_, 0, 0);
  k_pack_b<<<dim3((D_ * D_) / 256, 1, 1), blk, 0, stream>>>(
      attnW + D_, attnWeP, D_, D_, 2 * D_, 0, 0);

  // ---- encoder layer 0 ----
  k_gi0<<<dim3((2u * T_ * B_ * G3H) / 256), blk, 0, stream>>>(x, eWih0, ebih0, GIbuf);
  k_enc_recur<<<dim3(4, 2), blk, 0, stream>>>(GIbuf, whhP, ebhh0, seq1, hlastBuf);
  // ---- encoder layer 1 ---- (N=768 -> 48 tiles, 16 per block -> grid.y=3)
  k_gemm_bias<<<dim3(1024, 3, 2), blk, 0, stream>>>(
      seq1, D_, 0, wihP, (size_t)D_ * G3H, ebih, G3H, GIbuf, G3H,
      (size_t)T_ * B_ * G3H, 16);
  k_enc_recur<<<dim3(4, 2), blk, 0, stream>>>(
      GIbuf, whhP + (size_t)2 * H_ * G3H, ebhh, seq2, hlastBuf + (size_t)1 * 2 * B_ * H_);
  // ---- encoder layer 2 ----
  k_gemm_bias<<<dim3(1024, 3, 2), blk, 0, stream>>>(
      seq2, D_, 0, wihP + (size_t)2 * D_ * G3H, (size_t)D_ * G3H, ebih + 2 * G3H, G3H,
      GIbuf, G3H, (size_t)T_ * B_ * G3H, 16);
  k_enc_recur<<<dim3(4, 2), blk, 0, stream>>>(
      GIbuf, whhP + (size_t)4 * H_ * G3H, ebhh + 2 * G3H, seq1,
      hlastBuf + (size_t)2 * 2 * B_ * H_);
  // ---- attention projection (attn_b folded in; N=512 -> 32 tiles -> grid.y=2)
  k_gemm_bias<<<dim3(1024, 2, 1), blk, 0, stream>>>(
      seq1, D_, 0, attnWeP, 0, attnB, 0, projBuf, D_, 0, 16);
  // ---- decoder ----
  k_decoder<<<dim3(4), blk, 0, stream>>>(seq1, projBuf, hlastBuf, attnWhP, fcConP,
                                         fcConB, dWihP, dWhhP, dbih, dbhh, vvec, fcW,
                                         fcb, outp);
}
